// MAICAgent_34376918237822
// MI455X (gfx1250) — matrix operand order repaired and checked by reading, not compiled, not run on hardware
//
#include <hip/hip_runtime.h>
#include <hip/hip_bf16.h>
#include <math.h>

// ---------------------------------------------------------------------------
// Problem constants (from reference)
// ---------------------------------------------------------------------------
#define BSZ      4096
#define NROWS    (BSZ * 8)          // 32768
#define RROWS    (BSZ * 64)         // 262144
#define NACT     30
#define RNN_     64
#define INSHAPE  186

typedef __attribute__((ext_vector_type(16))) _Float16 v16h;
typedef __attribute__((ext_vector_type(8)))  _Float16 v8h;
typedef __attribute__((ext_vector_type(4)))  _Float16 v4h;
typedef __attribute__((ext_vector_type(8)))  float    v8f;

__device__ __forceinline__ float lrelu(float x) { return x > 0.f ? x : 0.01f * x; }

// ---------------------------------------------------------------------------
// gfx1250 async global->LDS staging (GLOBAL_LOAD_ASYNC_TO_LDS_B32, ASYNCcnt)
// ---------------------------------------------------------------------------
__device__ __forceinline__ unsigned lds_byte_off(void* p) {
  return (unsigned)(unsigned long long)(__attribute__((address_space(3))) void*)p;
}
__device__ __forceinline__ void async_load_f32(void* ldsDst, const float* gsrc) {
  asm volatile("global_load_async_to_lds_b32 %0, %1, off"
               :: "v"(lds_byte_off(ldsDst)), "v"((unsigned long long)(size_t)gsrc)
               : "memory");
}
__device__ __forceinline__ void async_wait0() {
  asm volatile("s_wait_asynccnt 0x0" ::: "memory");
}

// ---------------------------------------------------------------------------
// WMMA fragment helpers (CDNA5 16x16x32 f16 layout, wave32)
//   A (16x32): lane l: M=l&15; halves 0..7 -> K=(l<16?0:8)+i ; 8..15 -> +16
//   B (32x16): lane l: N=l&15; halves 0..7 -> K=(l<16?0:8)+i ; 8..15 -> +16
//   C/D:       vgpr r, lane l: row = r + (l<16?0:8), col = l&15
// B matrices are PRE-PACKED in fragment order -> one contiguous v16h per lane.
// ---------------------------------------------------------------------------
__device__ __forceinline__ v16h a_frag_f16_lds(const _Float16* As, int ld) {
  int l  = threadIdx.x & 31;
  int m  = l & 15;
  int b0 = (l & 16) ? 8 : 0;
  const _Float16* p = As + m * ld + b0;
  v8h lo = *(const v8h*)p;          // ds_load_b128
  v8h hi = *(const v8h*)(p + 16);   // ds_load_b128
  v16h a;
#pragma unroll
  for (int i = 0; i < 8; ++i) { a[i] = lo[i]; a[i + 8] = hi[i]; }
  return a;
}

__device__ __forceinline__ v16h b_frag_packed(const _Float16* Wp, int ntiles, int kt, int nt) {
  int lane = threadIdx.x & 31;
  return *(const v16h*)(Wp + ((size_t)(kt * ntiles + nt) * 32 + lane) * 16);  // 32B aligned
}

// ---------------------------------------------------------------------------
// Weight conversion: f32 -> f16, padded, packed in WMMA B-fragment order.
// dst[((kt*ntiles+nt)*32 + lane)*16 + i] = W[kt*32 + (lane&16)/2 + (i&7) + (i&8)*2][nt*16 + (lane&15)]
// ---------------------------------------------------------------------------
__global__ void convw_pack_kernel(const float* __restrict__ src, _Float16* __restrict__ dst,
                                  int K, int Ncol, int Kpad, int Npad,
                                  int srcStride, int colOff, int tr) {
  int idx = blockIdx.x * blockDim.x + threadIdx.x;
  if (idx >= Kpad * Npad) return;
  int i      = idx & 15;
  int lane   = (idx >> 4) & 31;
  int tile   = idx >> 9;
  int ntiles = Npad >> 4;
  int kt = tile / ntiles, nt = tile % ntiles;
  int k = kt * 32 + ((lane & 16) >> 1) + (i & 7) + ((i & 8) << 1);
  int n = nt * 16 + (lane & 15);
  float v = 0.f;
  if (k < K && n < Ncol)
    v = tr ? src[(size_t)n * srcStride + k]
           : src[(size_t)k * srcStride + colOff + n];
  dst[idx] = (_Float16)v;
}

// ---------------------------------------------------------------------------
// Generic GEMM: C[M x Nreal] = act(A[M x Kreal] @ W[Kpad x Npad] + bias)
// 128 thr = 4 waves, 16x64 tile/block. A slab streamed global->LDS with
// async-to-LDS (single divergence region: k column is per-thread invariant),
// converted once to an f16 tile (b128/b64 vector LDS ops), used by 4 waves.
// ---------------------------------------------------------------------------
__global__ void __launch_bounds__(128)
gemm_f16w(const float* __restrict__ A, int Kreal, int lda,
          const _Float16* __restrict__ Wp, int Kpad, int Npad,
          const float* __restrict__ bias, float* __restrict__ C,
          int ldc, int Nreal, int act) {
  __shared__ __align__(16) float    Af[16 * 32];
  __shared__ __align__(16) _Float16 Ah[16 * 40];
  const int row0   = blockIdx.x * 16;
  const int wave   = threadIdx.x >> 5;
  const int n0     = blockIdx.y * 64 + wave * 16;
  const int ntiles = Npad >> 4;
  const int kcol   = threadIdx.x & 31;   // staging column, loop-invariant
  const int rbase  = threadIdx.x >> 5;   // staging row base, loop-invariant
  v8f acc = {};
  for (int k0 = 0; k0 < Kpad; k0 += 32) {
    // async-stage 16x32 f32 slab into LDS (zero-pad past Kreal); one branch
    const int kg = k0 + kcol;
    if (kg < Kreal) {
#pragma unroll
      for (int j = 0; j < 4; ++j) {
        int r = rbase + 4 * j;
        async_load_f32(&Af[r * 32 + kcol], A + (size_t)(row0 + r) * lda + kg);
      }
    } else {
#pragma unroll
      for (int j = 0; j < 4; ++j) Af[(rbase + 4 * j) * 32 + kcol] = 0.f;
    }
    async_wait0();
    __syncthreads();
    // one-shot f32 -> f16 conversion: 4 contiguous elems/thread (b128 -> b64)
    {
      int e4 = threadIdx.x * 4;
      int r = e4 >> 5, k = e4 & 31;
      float4 f = *(const float4*)&Af[r * 32 + k];
      v4h hv;
      hv[0] = (_Float16)f.x; hv[1] = (_Float16)f.y;
      hv[2] = (_Float16)f.z; hv[3] = (_Float16)f.w;
      *(v4h*)&Ah[r * 40 + k] = hv;
    }
    __syncthreads();
    if (n0 < Npad) {
      v16h af = a_frag_f16_lds(Ah, 40);
      v16h bf = b_frag_packed(Wp, ntiles, k0 >> 5, n0 >> 4);
      acc = __builtin_amdgcn_wmma_f32_16x16x32_f16(false, af, false, bf,
                                                   (short)0, acc, false, false);
    }
    __syncthreads();
  }
  if (n0 < Npad) {
    int l = threadIdx.x & 31;
    int n = l & 15;
    int mb = (l < 16) ? 0 : 8;
    int col = n0 + n;
    if (col < Nreal) {
      float bv = bias ? bias[col] : 0.f;
#pragma unroll
      for (int r = 0; r < 8; ++r) {
        float v = acc[r] + bv;
        if (act) v = lrelu(v);
        C[(size_t)(row0 + mb + r) * ldc + col] = v;
      }
    }
  }
}

// ---------------------------------------------------------------------------
// Fused 2-layer MLP on virtual concat(h_rep[64], latent[16]) -> 256 -> 30
// One block = 16 rows; X and hidden live as f16 tiles in LDS; packed weights.
// ---------------------------------------------------------------------------
__global__ void __launch_bounds__(128)
mlp2_fused(const float* __restrict__ H,      // [NROWS, 64], row = r>>3
           const float* __restrict__ LATv,   // [RROWS, 16]
           const _Float16* __restrict__ W1p, // [96, 256] packed
           const float* __restrict__ b1,     // [256]
           const _Float16* __restrict__ W2p, // [256, 32] packed
           const float* __restrict__ b2,     // [30]
           float* __restrict__ OUT) {        // [RROWS, 32]
  __shared__ __align__(16) _Float16 Xh[16 * 104];
  __shared__ __align__(16) _Float16 Hs[16 * 272];
  const int row0 = blockIdx.x * 16;
  const int wave = threadIdx.x >> 5;
  // stage gathered X tile (16 x 96 + pad): cols 0..63 from H, 64..79 LAT, rest 0
  for (int e = threadIdx.x; e < 16 * 96; e += 128) {
    int r = e / 96, c = e % 96;
    int grow = row0 + r;
    float v;
    if (c < 64)      v = H[(size_t)(grow >> 3) * 64 + c];
    else if (c < 80) v = LATv[(size_t)grow * 16 + (c - 64)];
    else             v = 0.f;
    Xh[r * 104 + c] = (_Float16)v;
  }
  __syncthreads();
  // layer 1: each wave computes 4 hidden 16x16 tiles, K = 96
  for (int t = 0; t < 4; ++t) {
    int n0 = wave * 64 + t * 16;
    v8f acc = {};
#pragma unroll
    for (int k0 = 0; k0 < 96; k0 += 32) {
      v16h af = a_frag_f16_lds(Xh + k0, 104);
      v16h bf = b_frag_packed(W1p, 16, k0 >> 5, n0 >> 4);
      acc = __builtin_amdgcn_wmma_f32_16x16x32_f16(false, af, false, bf,
                                                   (short)0, acc, false, false);
    }
    int l = threadIdx.x & 31;
    int n = l & 15, mb = (l < 16) ? 0 : 8;
    int col = n0 + n;
    float bv = b1[col];
#pragma unroll
    for (int r = 0; r < 8; ++r)
      Hs[(mb + r) * 272 + col] = (_Float16)lrelu(acc[r] + bv);
  }
  __syncthreads();
  // layer 2: waves 0..1 compute the two output tiles, K = 256
  if (wave < 2) {
    int n0 = wave * 16;
    v8f acc = {};
#pragma unroll
    for (int k0 = 0; k0 < 256; k0 += 32) {
      v16h af = a_frag_f16_lds(Hs + k0, 272);
      v16h bf = b_frag_packed(W2p, 2, k0 >> 5, n0 >> 4);
      acc = __builtin_amdgcn_wmma_f32_16x16x32_f16(false, af, false, bf,
                                                   (short)0, acc, false, false);
    }
    int l = threadIdx.x & 31;
    int n = l & 15, mb = (l < 16) ? 0 : 8;
    int col = n0 + n;
    if (col < NACT) {
      float bv = b2[col];
#pragma unroll
      for (int r = 0; r < 8; ++r)
        OUT[(size_t)(row0 + mb + r) * 32 + col] = acc[r] + bv;
    }
  }
}

// ---------------------------------------------------------------------------
// Elementwise kernels
// ---------------------------------------------------------------------------
__global__ void zero_kernel(float* p, int n) {
  int i = blockIdx.x * blockDim.x + threadIdx.x;
  if (i < n) p[i] = 0.f;
}

__global__ void gru_kernel(const float* __restrict__ gi, const float* __restrict__ gh,
                           const float* __restrict__ h0, float* __restrict__ h) {
  int i = blockIdx.x * blockDim.x + threadIdx.x;
  if (i >= NROWS * RNN_) return;
  int row = i >> 6, j = i & 63;
  const float* GI = gi + (size_t)row * 192;
  const float* GH = gh + (size_t)row * 192;
  float r  = 1.f / (1.f + __expf(-(GI[j] + GH[j])));
  float z  = 1.f / (1.f + __expf(-(GI[64 + j] + GH[64 + j])));
  float ng = tanhf(GI[128 + j] + r * GH[128 + j]);
  h[i] = (1.f - z) * ng + z * h0[i];
}

__global__ void colstats_kernel(const float* __restrict__ Z, int rowsPerBlock,
                                float* __restrict__ sum, float* __restrict__ sq) {
  int c  = threadIdx.x;                 // 256 columns
  int r0 = blockIdx.x * rowsPerBlock;
  float s = 0.f, q = 0.f;
  for (int r = 0; r < rowsPerBlock; ++r) {
    float v = Z[(size_t)(r0 + r) * 256 + c];
    s += v; q += v * v;
  }
  atomicAdd(&sum[c], s);
  atomicAdd(&sq[c], q);
}

__global__ void bnapply_kernel(float* __restrict__ Z, const float* __restrict__ sum,
                               const float* __restrict__ sq, const float* __restrict__ g,
                               const float* __restrict__ b) {
  size_t i = (size_t)blockIdx.x * blockDim.x + threadIdx.x;
  if (i >= (size_t)NROWS * 256) return;
  int c = (int)(i & 255);
  float m = sum[c] * (1.f / NROWS);
  float v = sq[c] * (1.f / NROWS) - m * m;
  float x = (Z[i] - m) * rsqrtf(v + 1e-5f) * g[c] + b[c];
  Z[i] = lrelu(x);
}

__global__ void latent_kernel(const float* __restrict__ INF, const float* __restrict__ EPS,
                              float* __restrict__ LATo) {
  size_t i = (size_t)blockIdx.x * blockDim.x + threadIdx.x;
  if (i >= (size_t)NROWS * 128) return;
  size_t row = i >> 7;  int l = (int)(i & 127);
  float mu = INF[row * 256 + l];
  float lv = INF[row * 256 + 128 + l];
  float var = fmaxf(__expf(lv), 0.002f);
  LATo[i] = mu + sqrtf(var) * EPS[i];   // reshape (bs*n*n, L) is contiguous
}

// ---------------------------------------------------------------------------
// Fused attention: alpha = softmax(masked q.k/sqrt(ATT)); out = q0 + sum_i a*V
// One block per batch element b.
// ---------------------------------------------------------------------------
__global__ void __launch_bounds__(256)
combine_kernel(const float* __restrict__ Q,   // [NROWS,32]
               const float* __restrict__ Kr,  // [RROWS,32], row (b*8+a)*8+i
               const float* __restrict__ V,   // [RROWS,32], row (b*8+i)*8+a
               const float* __restrict__ q0,  // [NROWS,32]
               float* __restrict__ out) {     // [NROWS,30]
  __shared__ float al[64];
  int b = blockIdx.x, t = threadIdx.x;
  const float scale = 0.17677669529663687f;   // 1/sqrt(32)
  if (t < 64) {
    int i = t >> 3, a = t & 7;
    const float4* q = (const float4*)(Q  + (size_t)(b * 8 + i) * 32);
    const float4* k = (const float4*)(Kr + ((size_t)(b * 8 + a) * 8 + i) * 32);
    float s = 0.f;
#pragma unroll
    for (int e = 0; e < 8; ++e) {
      float4 qa = q[e], kb = k[e];
      s += qa.x * kb.x + qa.y * kb.y + qa.z * kb.z + qa.w * kb.w;
    }
    al[t] = (i == a) ? -1e9f : s * scale;
  }
  __syncthreads();
  if (t < 8) {                                // softmax over a, per i
    float mx = -1e30f;
#pragma unroll
    for (int a = 0; a < 8; ++a) mx = fmaxf(mx, al[t * 8 + a]);
    float den = 0.f;
#pragma unroll
    for (int a = 0; a < 8; ++a) { float e = __expf(al[t * 8 + a] - mx); al[t * 8 + a] = e; den += e; }
    float inv = 1.f / den;
#pragma unroll
    for (int a = 0; a < 8; ++a) al[t * 8 + a] *= inv;
  }
  __syncthreads();
  if (t < 8 * NACT) {                         // out[b,a,c] = q0 + sum_i al[i,a]*V[b,i,a,c]
    int a = t / NACT, c = t % NACT;
    float s = 0.f;
#pragma unroll
    for (int i = 0; i < 8; ++i)
      s += al[i * 8 + a] * V[((size_t)(b * 8 + i) * 8 + a) * 32 + c];
    out[(size_t)(b * 8 + a) * NACT + c] = q0[(size_t)(b * 8 + a) * 32 + c] + s;
  }
}

// ---------------------------------------------------------------------------
// Host launcher
// ---------------------------------------------------------------------------
extern "C" void kernel_launch(void* const* d_in, const int* in_sizes, int n_in,
                              void* d_out, int out_size, void* d_ws, size_t ws_size,
                              hipStream_t stream) {
  (void)in_sizes; (void)n_in; (void)out_size; (void)ws_size;
  // ---- inputs (JAX pytree order: dicts sorted by key) ----
  const float* inputs   = (const float*)d_in[0];
  const float* h0       = (const float*)d_in[1];
  const float* eps_inf  = (const float*)d_in[2];
  const float* eps_dem  = (const float*)d_in[3];
  const float* dem_bn_b = (const float*)d_in[4];
  const float* dem_bn_g = (const float*)d_in[5];
  const float* dem_l1_W = (const float*)d_in[6];
  const float* dem_l1_b = (const float*)d_in[7];
  const float* dem_l2_W = (const float*)d_in[8];
  const float* dem_l2_b = (const float*)d_in[9];
  const float* fc2_W    = (const float*)d_in[10];
  const float* fc2_b    = (const float*)d_in[11];
  const float* gv_l1_W  = (const float*)d_in[12];
  const float* gv_l1_b  = (const float*)d_in[13];
  const float* gv_l2_W  = (const float*)d_in[14];
  const float* gv_l2_b  = (const float*)d_in[15];
  const float* inf_bn_b = (const float*)d_in[16];
  const float* inf_bn_g = (const float*)d_in[17];
  const float* inf_l1_W = (const float*)d_in[18];
  const float* inf_l1_b = (const float*)d_in[19];
  const float* inf_l2_W = (const float*)d_in[20];
  const float* inf_l2_b = (const float*)d_in[21];
  const float* qkv_W    = (const float*)d_in[22];
  const float* qkv_b    = (const float*)d_in[23];
  const float* msg_l1_W = (const float*)d_in[24];
  const float* msg_l1_b = (const float*)d_in[25];
  const float* msg_l2_W = (const float*)d_in[26];
  const float* msg_l2_b = (const float*)d_in[27];
  const float* rnn_b_hh = (const float*)d_in[28];
  const float* rnn_b_ih = (const float*)d_in[29];
  const float* rnn_w_hh = (const float*)d_in[30];
  const float* rnn_w_ih = (const float*)d_in[31];
  const float* tark_W   = (const float*)d_in[32];
  const float* tark_b   = (const float*)d_in[33];
  const float* tarq_W   = (const float*)d_in[34];
  const float* tarq_b   = (const float*)d_in[35];
  const float* tiny_W   = (const float*)d_in[36];
  const float* tiny_b   = (const float*)d_in[37];
  const float* wkey_W   = (const float*)d_in[38];
  const float* wkey_b   = (const float*)d_in[39];
  const float* wq_W     = (const float*)d_in[40];
  const float* wq_b     = (const float*)d_in[41];

  float* out = (float*)d_out;
  float* ret_q  = out;
  float* ret_gq = out + (size_t)NROWS * NACT;
  float* hbuf   = out + (size_t)2 * NROWS * NACT;   // h output, reused as GEMM input

  // ---- workspace carve-up ----
  char* ws = (char*)d_ws;
  size_t off = 0;
  auto ALLOC = [&](size_t bytes) { char* p = ws + off; off = (off + bytes + 255) & ~(size_t)255; return p; };
  _Float16* wWv   = (_Float16*)ALLOC(192 * 64  * 2);
  _Float16* wIh   = (_Float16*)ALLOC(64  * 192 * 2);
  _Float16* wHh   = (_Float16*)ALLOC(64  * 192 * 2);
  _Float16* wFc2  = (_Float16*)ALLOC(64  * 32  * 2);
  _Float16* wTiny = (_Float16*)ALLOC(64  * 16  * 2);
  _Float16* wQ    = (_Float16*)ALLOC(64  * 32  * 2);
  _Float16* wTarQ = (_Float16*)ALLOC(64  * 32  * 2);
  _Float16* wInf1 = (_Float16*)ALLOC(32  * 256 * 2);
  _Float16* wInf2 = (_Float16*)ALLOC(256 * 256 * 2);
  _Float16* wDem1 = (_Float16*)ALLOC(64  * 256 * 2);
  _Float16* wDem2 = (_Float16*)ALLOC(256 * 256 * 2);
  _Float16* wMsg1 = (_Float16*)ALLOC(96  * 256 * 2);
  _Float16* wMsg2 = (_Float16*)ALLOC(256 * 32  * 2);
  _Float16* wGv1  = (_Float16*)ALLOC(96  * 256 * 2);
  _Float16* wGv2  = (_Float16*)ALLOC(256 * 32  * 2);
  _Float16* wKey  = (_Float16*)ALLOC(32  * 32  * 2);
  _Float16* wTarK = (_Float16*)ALLOC(32  * 32  * 2);
  float* stats   = (float*)ALLOC(4 * 256 * 4);      // sums_i, sq_i, sums_d, sq_d
  float* feat    = (float*)ALLOC((size_t)NROWS * 64  * 4);
  float* gi      = (float*)ALLOC((size_t)NROWS * 192 * 4);
  float* gh      = (float*)ALLOC((size_t)NROWS * 192 * 4);
  float* q_org   = (float*)ALLOC((size_t)NROWS * 32  * 4);
  float* tinyv   = (float*)ALLOC((size_t)NROWS * 16  * 4);
  float* queryv  = (float*)ALLOC((size_t)NROWS * 32  * 4);
  float* gqueryv = (float*)ALLOC((size_t)NROWS * 32  * 4);
  float* z1i     = (float*)ALLOC((size_t)NROWS * 256 * 4);
  float* inferv  = (float*)ALLOC((size_t)NROWS * 256 * 4);
  float* z1d     = (float*)ALLOC((size_t)NROWS * 256 * 4);
  float* demv    = (float*)ALLOC((size_t)NROWS * 256 * 4);
  float* lat_i   = (float*)ALLOC((size_t)RROWS * 16  * 4);
  float* lat_d   = (float*)ALLOC((size_t)RROWS * 16  * 4);
  float* key_i   = (float*)ALLOC((size_t)RROWS * 32  * 4);
  float* key_d   = (float*)ALLOC((size_t)RROWS * 32  * 4);
  float* msgbuf  = (float*)ALLOC((size_t)RROWS * 32  * 4);
  float* gvbuf   = (float*)ALLOC((size_t)RROWS * 32  * 4);

  float* sum_i = stats;        float* sq_i = stats + 256;
  float* sum_d = stats + 512;  float* sq_d = stats + 768;

  auto conv = [&](const float* src, _Float16* dst, int K, int Ncol, int Kpad, int Npad,
                  int stride, int colOff, int tr) {
    int tot = Kpad * Npad;
    convw_pack_kernel<<<(tot + 255) / 256, 256, 0, stream>>>(src, dst, K, Ncol, Kpad, Npad, stride, colOff, tr);
  };
  auto gemm = [&](const float* A, int M, int Kreal, int lda, const _Float16* Wp,
                  int Kpad, int Npad, const float* bias, float* C, int ldc, int Nreal, int act) {
    dim3 grid(M / 16, (Npad + 63) / 64);
    gemm_f16w<<<grid, 128, 0, stream>>>(A, Kreal, lda, Wp, Kpad, Npad, bias, C, ldc, Nreal, act);
  };

  // ---- 0) zero BN stat accumulators ----
  zero_kernel<<<4, 256, 0, stream>>>(stats, 1024);

  // ---- 1) weight conversion to packed f16 fragments ----
  conv(qkv_W,    wWv,   INSHAPE, 64,  192, 64,  192, 128, 0);  // V slice only
  conv(rnn_w_ih, wIh,   64, 192, 64, 192, 64, 0, 1);           // transpose
  conv(rnn_w_hh, wHh,   64, 192, 64, 192, 64, 0, 1);           // transpose
  conv(fc2_W,    wFc2,  64, 30,  64, 32,  30, 0, 0);
  conv(tiny_W,   wTiny, 64, 16,  64, 16,  16, 0, 0);
  conv(wq_W,     wQ,    64, 32,  64, 32,  32, 0, 0);
  conv(tarq_W,   wTarQ, 64, 32,  64, 32,  32, 0, 0);
  conv(inf_l1_W, wInf1, 16, 256, 32, 256, 256, 0, 0);
  conv(inf_l2_W, wInf2, 256, 256, 256, 256, 256, 0, 0);
  conv(dem_l1_W, wDem1, 64, 256, 64, 256, 256, 0, 0);
  conv(dem_l2_W, wDem2, 256, 256, 256, 256, 256, 0, 0);
  conv(msg_l1_W, wMsg1, 80, 256, 96, 256, 256, 0, 0);
  conv(msg_l2_W, wMsg2, 256, 30, 256, 32, 30, 0, 0);
  conv(gv_l1_W,  wGv1,  80, 256, 96, 256, 256, 0, 0);
  conv(gv_l2_W,  wGv2,  256, 30, 256, 32, 30, 0, 0);
  conv(wkey_W,   wKey,  16, 32, 32, 32, 32, 0, 0);
  conv(tark_W,   wTarK, 16, 32, 32, 32, 32, 0, 0);

  // ---- 2) feat = inputs @ Wv + bv ; GRU gates ; h ----
  gemm(inputs, NROWS, INSHAPE, INSHAPE, wWv, 192, 64, qkv_b + 128, feat, 64, 64, 0);
  gemm(feat, NROWS, 64, 64, wIh, 64, 192, rnn_b_ih, gi, 192, 192, 0);
  gemm(h0,   NROWS, 64, 64, wHh, 64, 192, rnn_b_hh, gh, 192, 192, 0);
  gru_kernel<<<(NROWS * RNN_ + 255) / 256, 256, 0, stream>>>(gi, gh, h0, hbuf);

  // ---- 3) heads off h ----
  gemm(hbuf, NROWS, 64, 64, wFc2,  64, 32, fc2_b,  q_org,   32, 30, 0);
  gemm(hbuf, NROWS, 64, 64, wTiny, 64, 16, tiny_b, tinyv,   16, 16, 0);
  gemm(hbuf, NROWS, 64, 64, wQ,    64, 32, wq_b,   queryv,  32, 32, 0);
  gemm(hbuf, NROWS, 64, 64, wTarQ, 64, 32, tarq_b, gqueryv, 32, 32, 0);

  // ---- 4) inference MLP with batch-norm (2-pass stats) ----
  gemm(tinyv, NROWS, 16, 16, wInf1, 32, 256, inf_l1_b, z1i, 256, 256, 0);
  colstats_kernel<<<128, 256, 0, stream>>>(z1i, NROWS / 128, sum_i, sq_i);
  bnapply_kernel<<<(NROWS * 256 + 255) / 256, 256, 0, stream>>>(z1i, sum_i, sq_i, inf_bn_g, inf_bn_b);
  gemm(z1i, NROWS, 256, 256, wInf2, 256, 256, inf_l2_b, inferv, 256, 256, 0);
  latent_kernel<<<(NROWS * 128 + 255) / 256, 256, 0, stream>>>(inferv, eps_inf, lat_i);

  // ---- 5) demand MLP with batch-norm ----
  gemm(hbuf, NROWS, 64, 64, wDem1, 64, 256, dem_l1_b, z1d, 256, 256, 0);
  colstats_kernel<<<128, 256, 0, stream>>>(z1d, NROWS / 128, sum_d, sq_d);
  bnapply_kernel<<<(NROWS * 256 + 255) / 256, 256, 0, stream>>>(z1d, sum_d, sq_d, dem_bn_g, dem_bn_b);
  gemm(z1d, NROWS, 256, 256, wDem2, 256, 256, dem_l2_b, demv, 256, 256, 0);
  latent_kernel<<<(NROWS * 128 + 255) / 256, 256, 0, stream>>>(demv, eps_dem, lat_d);

  // ---- 6) attention keys ----
  gemm(lat_i, RROWS, 16, 16, wKey,  32, 32, wkey_b, key_i, 32, 32, 0);
  gemm(lat_d, RROWS, 16, 16, wTarK, 32, 32, tark_b, key_d, 32, 32, 0);

  // ---- 7) fused msg / g_value MLPs on 262144 pair-rows ----
  mlp2_fused<<<RROWS / 16, 128, 0, stream>>>(hbuf, lat_i, wMsg1, msg_l1_b, wMsg2, msg_l2_b, msgbuf);
  mlp2_fused<<<RROWS / 16, 128, 0, stream>>>(hbuf, lat_d, wGv1,  gv_l1_b,  wGv2,  gv_l2_b,  gvbuf);

  // ---- 8) attention softmax + weighted combine -> outputs ----
  combine_kernel<<<BSZ, 256, 0, stream>>>(queryv,  key_i, msgbuf, q_org, ret_q);
  combine_kernel<<<BSZ, 256, 0, stream>>>(gqueryv, key_d, gvbuf,  q_org, ret_gq);
  // h was written directly into d_out by gru_kernel
}
